// Multi_head_GATLayer_44461501448906
// MI455X (gfx1250) — matrix operand order repaired and checked by reading, not compiled, not run on hardware
//
#include <hip/hip_runtime.h>
#include <hip/hip_bf16.h>

// ---------------------------------------------------------------------------
// Multi-head GAT layer, fused flash-attention style, bf16 WMMA on gfx1250.
// N=4096, IN_F=128, OUT_F=64, H=8.
//
// Pipeline:
//   K0  zero accumulators (denacc + numacc)
//   K1  XpT[c][n] = sum_k Wp[c,k]*A[n,k]    (bf16 out, v_wmma_f32_16x16x32_bf16)
//   K2  ss[h][n], st[h][n] (dot of XpT column with a_src/a_tgt)
//   K2b stmax[h] = max_n st[h][n]   (exact per-row softmax max via monotonicity)
//   K3  per (16-query tile, head, m-slice): stream m in chunks of 32,
//       P^T = exp(lrelu(ss+st)+mask - rmax) as WMMA B operand (bf16),
//       Xp^T as WMMA A operand, accumulate out^T (f32) + denom; atomicAdd.
//   K4  out[n][f] = relu( (1/8) * sum_h num[h][f][n] / den[h][n] )
// ---------------------------------------------------------------------------

#define GN     4096
#define GIN    128
#define GOUT   64
#define GH     8
#define GC     (GH*GOUT)      // 512 projected channels

typedef __attribute__((ext_vector_type(16))) __bf16 v16bf;
typedef __attribute__((ext_vector_type(8)))  __bf16 v8bf;
typedef __attribute__((ext_vector_type(8)))  float  v8f;

__device__ __forceinline__ float lrelu02(float x) {
    return x > 0.0f ? x : 0.2f * x;
}

// ---------------- K0: zero a float region ----------------------------------
__global__ void gat_zero_kernel(float* __restrict__ p, int n) {
    int t = blockIdx.x * blockDim.x + threadIdx.x;
    if (t < n) p[t] = 0.0f;
}

// ---------------- K1: projection GEMM, output transposed bf16 ---------------
// XpT[c][n] = sum_k Wp[c,k] * A[n,k]   (XpT: [512][4096] bf16)
// One wave computes a 16(c) x 16(n) tile; K=128 -> 4 WMMA(16x16x32).
__global__ __launch_bounds__(32)
void gat_proj_kernel(const float* __restrict__ A,   // [4096][128]
                     const float* __restrict__ Wp,  // [512][128]
                     __bf16* __restrict__ XpT) {    // [512][4096]
    const int nbase = blockIdx.x * 16;
    const int cbase = blockIdx.y * 16;
    const int lane  = threadIdx.x;
    const int lo    = lane & 15;
    const int hi    = lane >> 4;

    v8f acc = {};
    #pragma unroll
    for (int k0 = 0; k0 < GIN; k0 += 32) {
        v16bf a, b;
        // A-operand (16x32, MxK): row c = cbase+lo; k = k0 + hi*16 + i  (contiguous)
        const float* wrow = Wp + (size_t)(cbase + lo) * GIN + k0 + hi * 16;
        #pragma unroll
        for (int i = 0; i < 16; ++i)
            a[i] = (__bf16)wrow[i];
        // B-operand (32x16, KxN): col n = nbase+lo; k = k0 + hi*16 + i  (contiguous)
        const float* arow = A + (size_t)(nbase + lo) * GIN + k0 + hi * 16;
        #pragma unroll
        for (int i = 0; i < 16; ++i)
            b[i] = (__bf16)arow[i];
        acc = __builtin_amdgcn_wmma_f32_16x16x32_bf16(
                  false, a, false, b, (short)0, acc, false, false);
    }
    // C/D layout: VGPR r -> row c = cbase + r + 8*hi, col n = nbase + lo
    #pragma unroll
    for (int r = 0; r < 8; ++r)
        XpT[(size_t)(cbase + r + hi * 8) * GN + nbase + lo] = (__bf16)acc[r];
}

// ---------------- K2: ss / st vectors ---------------------------------------
__global__ void gat_scores_kernel(const __bf16* __restrict__ XpT,  // [512][4096]
                                  const float* __restrict__ a_src, // [8][64]
                                  const float* __restrict__ a_tgt, // [8][64]
                                  float* __restrict__ ss,          // [8][4096]
                                  float* __restrict__ st) {        // [8][4096]
    int t = blockIdx.x * blockDim.x + threadIdx.x;      // 32768 threads
    if (t >= GH * GN) return;
    int h = t >> 12;
    int n = t & (GN - 1);
    const __bf16* col = XpT + (size_t)h * GOUT * GN + n;
    float s0 = 0.0f, s1 = 0.0f;
    #pragma unroll 8
    for (int f = 0; f < GOUT; ++f) {
        float x = (float)col[(size_t)f * GN];
        s0 += x * a_src[h * GOUT + f];
        s1 += x * a_tgt[h * GOUT + f];
    }
    ss[h * GN + n] = s0;
    st[h * GN + n] = s1;
}

// ---------------- K2b: per-head max of st -----------------------------------
__global__ void gat_stmax_kernel(const float* __restrict__ st, float* __restrict__ stmax) {
    __shared__ float red[128];
    int h = blockIdx.x;
    float m = -3.0e38f;
    for (int n = threadIdx.x; n < GN; n += 128)
        m = fmaxf(m, st[h * GN + n]);
    red[threadIdx.x] = m;
    __syncthreads();
    for (int s = 64; s > 0; s >>= 1) {
        if (threadIdx.x < s) red[threadIdx.x] = fmaxf(red[threadIdx.x], red[threadIdx.x + s]);
        __syncthreads();
    }
    if (threadIdx.x == 0) stmax[h] = red[0];
}

// ---------------- K3: fused attention (flash-style, WMMA) -------------------
// grid: (256 ntiles, 8 heads, 4 m-slices), block = 1 wave (32 threads).
// out^T(16f x 16n) += Xp^T(16f x 32m) x P^T(32m x 16n), unnormalized;
// denominators accumulated separately; combined exactly in K4 since the
// row-max (rmax) is global and exact: rmax = lrelu(ss[n] + stmax[h]).
__global__ __launch_bounds__(32)
void gat_attn_kernel(const __bf16* __restrict__ XpT,   // [512][4096]
                     const float*  __restrict__ ss,    // [8][4096]
                     const float*  __restrict__ st,    // [8][4096]
                     const float*  __restrict__ stmax, // [8]
                     const float*  __restrict__ mask,  // [4096][4096]
                     float* __restrict__ numacc,       // [512][4096]
                     float* __restrict__ denacc) {     // [8][4096]
    const int n0     = blockIdx.x * 16;
    const int h      = blockIdx.y;
    const int mstart = blockIdx.z * (GN / 4);
    const int lane   = threadIdx.x;
    const int lo     = lane & 15;
    const int hi     = lane >> 4;

    const int   nq   = n0 + lo;
    const float ssv  = ss[h * GN + nq];
    const float rmax = lrelu02(ssv + stmax[h]);

    const __bf16* xph = XpT + (size_t)h * GOUT * GN;
    const float*  sth = st + h * GN;

    v8f   acc[4] = {};
    float den = 0.0f;

    for (int m0 = mstart; m0 < mstart + GN / 4; m0 += 32) {
        // ---- B operand: P^T (32m x 16n). lane covers query nq, m = m0+hi*16+i.
        v16bf b;
        const float4* stv = (const float4*)(sth + m0 + hi * 16);
        const float4* mkv = (const float4*)(mask + (size_t)nq * GN + m0 + hi * 16);
        #pragma unroll
        for (int q = 0; q < 4; ++q) {
            float4 sv = stv[q];
            float4 mv = mkv[q];
            float p0 = __expf(lrelu02(ssv + sv.x) + mv.x - rmax);
            float p1 = __expf(lrelu02(ssv + sv.y) + mv.y - rmax);
            float p2 = __expf(lrelu02(ssv + sv.z) + mv.z - rmax);
            float p3 = __expf(lrelu02(ssv + sv.w) + mv.w - rmax);
            den += p0 + p1 + p2 + p3;
            b[q * 4 + 0] = (__bf16)p0;
            b[q * 4 + 1] = (__bf16)p1;
            b[q * 4 + 2] = (__bf16)p2;
            b[q * 4 + 3] = (__bf16)p3;
        }
        // ---- A operands: Xp^T (16f x 32m), 4 feature tiles of 16 rows.
        #pragma unroll
        for (int j = 0; j < 4; ++j) {
            const __bf16* ap = xph + (size_t)(j * 16 + lo) * GN + m0 + hi * 16;
            v8bf r0 = *(const v8bf*)ap;          // k = hi*16 + 0..7
            v8bf r1 = *(const v8bf*)(ap + 8);    // k = hi*16 + 8..15
            v16bf a;
            #pragma unroll
            for (int i = 0; i < 8; ++i) { a[i] = r0[i]; a[i + 8] = r1[i]; }
            acc[j] = __builtin_amdgcn_wmma_f32_16x16x32_bf16(
                         false, a, false, b, (short)0, acc[j], false, false);
        }
    }

    // denominator: lanes n and n+16 each hold half of query n's sum; both add.
    atomicAdd(&denacc[h * GN + nq], den);
    // numerator: D VGPR r -> row f = j*16 + r + 8*hi, col n = lo (coalesced adds)
    #pragma unroll
    for (int j = 0; j < 4; ++j)
        #pragma unroll
        for (int r = 0; r < 8; ++r)
            atomicAdd(&numacc[(size_t)(h * GOUT + j * 16 + r + hi * 8) * GN + n0 + lo],
                      acc[j][r]);
}

// ---------------- K4: finalize (divide, head-mean, relu) --------------------
__global__ void gat_final_kernel(const float* __restrict__ numacc, // [512][4096]
                                 const float* __restrict__ denacc, // [8][4096]
                                 float* __restrict__ out) {        // [4096][64]
    int t = blockIdx.x * blockDim.x + threadIdx.x;   // 262144 threads, n-fast
    if (t >= GN * GOUT) return;
    int n = t & (GN - 1);
    int f = t >> 12;
    float s = 0.0f;
    #pragma unroll
    for (int h = 0; h < GH; ++h) {
        float d = fmaxf(denacc[h * GN + n], 1e-20f);
        s += numacc[(size_t)(h * GOUT + f) * GN + n] / d;
    }
    s *= (1.0f / (float)GH);
    out[(size_t)n * GOUT + f] = fmaxf(s, 0.0f);
}

// ---------------------------------------------------------------------------
extern "C" void kernel_launch(void* const* d_in, const int* in_sizes, int n_in,
                              void* d_out, int out_size, void* d_ws, size_t ws_size,
                              hipStream_t stream) {
    const float* inA   = (const float*)d_in[0];  // [4096][128]
    const float* mask  = (const float*)d_in[1];  // [4096][4096]
    const float* Wp    = (const float*)d_in[2];  // [512][128]
    const float* a_src = (const float*)d_in[3];  // [8][64]
    const float* a_tgt = (const float*)d_in[4];  // [8][64]
    float* out = (float*)d_out;                  // [4096][64]

    char* ws = (char*)d_ws;
    // workspace layout (bytes), all 256B aligned:
    __bf16* XpT    = (__bf16*)(ws + 0);                 //  4,194,304  [512][4096] bf16
    float*  ss     = (float*)(ws + 4194304);            //    131,072  [8][4096]
    float*  st     = (float*)(ws + 4325376);            //    131,072  [8][4096]
    float*  stmax  = (float*)(ws + 4456448);            //        256
    float*  denacc = (float*)(ws + 4456704);            //    131,072  [8][4096]
    float*  numacc = (float*)(ws + 4587776);            //  8,388,608  [512][4096]
    // total: 12,976,384 bytes

    // K0: zero denacc+numacc (contiguous region)
    {
        int nz = (GH * GN) + (GC * GN);   // 2,129,920 floats
        gat_zero_kernel<<<(nz + 255) / 256, 256, 0, stream>>>(denacc, nz);
    }
    // K1: projection (transposed bf16 output)
    gat_proj_kernel<<<dim3(GN / 16, GC / 16), 32, 0, stream>>>(inA, Wp, XpT);
    // K2: ss / st
    gat_scores_kernel<<<(GH * GN) / 256, 256, 0, stream>>>(XpT, a_src, a_tgt, ss, st);
    // K2b: per-head max of st
    gat_stmax_kernel<<<GH, 128, 0, stream>>>(st, stmax);
    // K3: fused attention
    gat_attn_kernel<<<dim3(GN / 16, GH, 4), 32, 0, stream>>>(
        XpT, ss, st, stmax, mask, numacc, denacc);
    // K4: finalize
    gat_final_kernel<<<(GN * GOUT) / 256, 256, 0, stream>>>(numacc, denacc, out);
}
